// SubspaceAwareGNNLayer_39496519254191
// MI455X (gfx1250) — matrix operand order repaired and checked by reading, not compiled, not run on hardware
//
#include <hip/hip_runtime.h>
#include <cstdint>

#define NN   8192
#define INF  512
#define OUTF 512
#define NSUB 8

#if defined(__has_builtin)
#if __has_builtin(__builtin_amdgcn_sched_barrier)
#define SCHED_FENCE() __builtin_amdgcn_sched_barrier(0)
#endif
#endif
#ifndef SCHED_FENCE
#define SCHED_FENCE()
#endif

typedef __attribute__((ext_vector_type(16))) _Float16     v16h;
typedef __attribute__((ext_vector_type(8)))  float        v8f;
typedef __attribute__((ext_vector_type(8)))  unsigned int v8u;
typedef __attribute__((ext_vector_type(4)))  unsigned int u32x4;
typedef __attribute__((ext_vector_type(2)))  unsigned int u32x2;
typedef __attribute__((ext_vector_type(4)))  float        f32x4;
typedef __attribute__((ext_vector_type(4)))  int          i32x4;

static __device__ __forceinline__ unsigned pkh2(float a, float b) {
  unsigned short ua = __builtin_bit_cast(unsigned short, (_Float16)a);
  unsigned short ub = __builtin_bit_cast(unsigned short, (_Float16)b);
  return (unsigned)ua | ((unsigned)ub << 16);
}

// ---------------------------------------------------------------------------
// K0: WT[s][n][k] (fp16) = W[s][k][n] (fp32).  8*512*512 elements.
// ---------------------------------------------------------------------------
__global__ void __launch_bounds__(256) wt_transpose_kernel(
    const float* __restrict__ W, _Float16* __restrict__ WT) {
  int idx = blockIdx.x * 256 + threadIdx.x;          // 0 .. 2^21-1
  int s = idx >> 18;
  int r = idx & ((1 << 18) - 1);
  int k = r >> 9;
  int n = r & 511;
  WT[(s << 18) + (n << 9) + k] = (_Float16)W[idx];
}

// ---------------------------------------------------------------------------
// K1: supportT[n][m] = (X @ W[label[m]])[m][n], fp16 out, f32 accumulate.
// Block = 256 thr (8 waves): 16 rows x all 512 cols.  Per-subspace row
// masking of the A fragment gives exact per-row weight selection.
// X fragments are software-pipelined; B loads are fenced away from WMMAs.
// ---------------------------------------------------------------------------
__global__ void __launch_bounds__(256) support_kernel(
    const float* __restrict__ X, const int* __restrict__ lab,
    const _Float16* __restrict__ WT, _Float16* __restrict__ supT) {
  const int MB   = blockIdx.x * 16;
  const int wid  = threadIdx.x >> 5;
  const int lane = threadIdx.x & 31;
  const int l16  = lane & 15;
  const int half = lane >> 4;
  const int NB0  = wid * 64;

  __shared__ int presence;
  if (threadIdx.x == 0) {
    int m = 0;
#pragma unroll
    for (int i = 0; i < 16; i++) m |= 1 << lab[MB + i];
    presence = m;
  }
  __syncthreads();
  const int pres  = presence;
  const int myrow = MB + l16;
  const int mylab = lab[myrow];

  v8f acc[4] = {};
  const v8u zerov = (v8u)0u;

  // prime X-fragment pipeline (lane row = l16; K = g*16 + half*8 + 0..7)
  const float* xrow = X + (size_t)myrow * INF + half * 8;
  f32x4 c0 = *(const f32x4*)(xrow);
  f32x4 c1 = *(const f32x4*)(xrow + 4);
  f32x4 c2 = *(const f32x4*)(xrow + 16);
  f32x4 c3 = *(const f32x4*)(xrow + 20);

  for (int kk = 0; kk < INF; kk += 32) {
    // prefetch next K-step's X fragment (wrap: harmless reload of step 0)
    int kn = (kk + 32 < INF) ? (kk + 32) : 0;
    f32x4 n0 = *(const f32x4*)(xrow + kn);
    f32x4 n1 = *(const f32x4*)(xrow + kn + 4);
    f32x4 n2 = *(const f32x4*)(xrow + kn + 16);
    f32x4 n3 = *(const f32x4*)(xrow + kn + 20);
    SCHED_FENCE();

    v8u araw;
    araw[0] = pkh2(c0[0], c0[1]); araw[1] = pkh2(c0[2], c0[3]);
    araw[2] = pkh2(c1[0], c1[1]); araw[3] = pkh2(c1[2], c1[3]);
    araw[4] = pkh2(c2[0], c2[1]); araw[5] = pkh2(c2[2], c2[3]);
    araw[6] = pkh2(c3[0], c3[1]); araw[7] = pkh2(c3[2], c3[3]);

#pragma unroll
    for (int s = 0; s < NSUB; s++) {
      if (!((pres >> s) & 1)) continue;        // block-uniform branch
      const _Float16* wb = WT + ((size_t)s << 18) + kk + half * 8;
      // issue all 4 B fragments, then fence: keeps them in distinct VGPRs
      // with staggered loadcnt waits instead of wait-0 before every WMMA.
      v8u b[4];
#pragma unroll
      for (int ng = 0; ng < 4; ng++) {
        int n = NB0 + ng * 16 + l16;
        b[ng] = __builtin_shufflevector(*(const u32x4*)(wb + ((size_t)n << 9)), *(const u32x4*)(wb + ((size_t)n << 9) + 16), 0, 1, 2, 3, 4, 5, 6, 7);
      }
      SCHED_FENCE();
      v8u am = (mylab == s) ? araw : zerov;    // zero non-member rows
      v16h av = __builtin_bit_cast(v16h, am);
#pragma unroll
      for (int ng = 0; ng < 4; ng++) {
        v16h bv = __builtin_bit_cast(v16h, b[ng]);
        acc[ng] = __builtin_amdgcn_wmma_f32_16x16x32_f16(
            false, av, false, bv, (short)0, acc[ng], false, false);
      }
    }
    c0 = n0; c1 = n1; c2 = n2; c3 = n3;
  }

  // D tile: lane col n = l16; VGPR r holds m = MB + 8*half + r -> pack 8
  // consecutive m as fp16: one 16B contiguous store.
#pragma unroll
  for (int ng = 0; ng < 4; ng++) {
    int n = NB0 + ng * 16 + l16;
    u32x4 o;
    o[0] = pkh2(acc[ng][0], acc[ng][1]);
    o[1] = pkh2(acc[ng][2], acc[ng][3]);
    o[2] = pkh2(acc[ng][4], acc[ng][5]);
    o[3] = pkh2(acc[ng][6], acc[ng][7]);
    *(u32x4*)(supT + (size_t)n * NN + MB + half * 8) = o;
  }
}

// ---------------------------------------------------------------------------
// K2: out = (adj .* same_label) @ support.  Block = 256 thr (8 waves) per
// 16-row tile covering all 512 cols -> adj streamed from HBM exactly once.
// Pipeline per chunk: [16 B-frag loads] | [adj+label prefetch for i+1] |
// [mask/pack/LDS-store chunk i] -> barrier -> [4x ds_load_b128] -> 8 WMMAs.
// ---------------------------------------------------------------------------
#define KCH 64
#define LDSROW 72   // 64 + 8 fp16 pad (kills ds_load_b128 bank conflicts)

__global__ void __launch_bounds__(256) gnn_kernel(
    const float* __restrict__ adj, const int* __restrict__ lab,
    const _Float16* __restrict__ supT, float* __restrict__ out) {
  __shared__ __align__(16) _Float16 abuf[2][16 * LDSROW];

  const int MB   = blockIdx.x * 16;
  const int wid  = threadIdx.x >> 5;
  const int lane = threadIdx.x & 31;
  const int l16  = lane & 15;
  const int half = lane >> 4;
  const int NB0  = wid * 64;

  // staging geometry: thread t covers adj[MB + t/16][kk + (t%16)*4 .. +3]
  const int srow = threadIdx.x >> 4;
  const int scol = (threadIdx.x & 15) * 4;
  const int slab = lab[MB + srow];
  const float* arow = adj + (size_t)(MB + srow) * NN + scol;

  v8f acc[4] = {};
  int p = 0;

  // prime the staging pipeline with chunk 0
  f32x4 a4 = *(const f32x4*)(arow);
  i32x4 lb = *(const i32x4*)(lab + scol);

  for (int kk = 0; kk < NN; kk += KCH) {
    // ---- issue all B fragments for this chunk (oldest loads) ----
    const _Float16* sb0 = supT + kk + half * 8;
    v8u bfr[8];
#pragma unroll
    for (int ks2 = 0; ks2 < 2; ks2++)
#pragma unroll
      for (int ng = 0; ng < 4; ng++) {
        int n = NB0 + ng * 16 + l16;
        bfr[ks2 * 4 + ng] = __builtin_shufflevector(*(const u32x4*)(sb0 + ks2 * 32 + (size_t)n * NN), *(const u32x4*)(sb0 + ks2 * 32 + (size_t)n * NN + 16), 0, 1, 2, 3, 4, 5, 6, 7);
      }
    SCHED_FENCE();

    // ---- prefetch adj + labels for next chunk (youngest loads) ----
    int kn = (kk + KCH < NN) ? (kk + KCH) : 0;   // wrap: harmless reload
    f32x4 a4n = *(const f32x4*)(arow + kn);
    i32x4 lbn = *(const i32x4*)(lab + kn + scol);
    SCHED_FENCE();

    // ---- mask + pack + LDS-store this chunk's adj tile ----
    float e0 = (lb[0] == slab) ? a4[0] : 0.f;
    float e1 = (lb[1] == slab) ? a4[1] : 0.f;
    float e2 = (lb[2] == slab) ? a4[2] : 0.f;
    float e3 = (lb[3] == slab) ? a4[3] : 0.f;
    u32x2 pk;
    pk[0] = pkh2(e0, e1);
    pk[1] = pkh2(e2, e3);
    *(u32x2*)&abuf[p][srow * LDSROW + scol] = pk;
    __syncthreads();

    // ---- A fragments for both K-substeps from LDS ----
    const _Float16* lrow = &abuf[p][l16 * LDSROW + half * 8];
    u32x4 a0lo = *(const u32x4*)(lrow);        // ks=0 : K = half*8    ..+7
    u32x4 a0hi = *(const u32x4*)(lrow + 16);   // ks=0 : K = 16+half*8 ..+7
    u32x4 a1lo = *(const u32x4*)(lrow + 32);   // ks=32: K = 32+half*8 ..+7
    u32x4 a1hi = *(const u32x4*)(lrow + 48);   // ks=32: K = 48+half*8 ..+7
    SCHED_FENCE();
    v8u am0, am1;
    am0[0] = a0lo[0]; am0[1] = a0lo[1]; am0[2] = a0lo[2]; am0[3] = a0lo[3];
    am0[4] = a0hi[0]; am0[5] = a0hi[1]; am0[6] = a0hi[2]; am0[7] = a0hi[3];
    am1[0] = a1lo[0]; am1[1] = a1lo[1]; am1[2] = a1lo[2]; am1[3] = a1lo[3];
    am1[4] = a1hi[0]; am1[5] = a1hi[1]; am1[6] = a1hi[2]; am1[7] = a1hi[3];
    v16h av0 = __builtin_bit_cast(v16h, am0);
    v16h av1 = __builtin_bit_cast(v16h, am1);

    // ---- 8 WMMAs; waits on bfr[] are staggered ----
#pragma unroll
    for (int ng = 0; ng < 4; ng++) {
      v16h bv = __builtin_bit_cast(v16h, bfr[ng]);
      acc[ng] = __builtin_amdgcn_wmma_f32_16x16x32_f16(
          false, av0, false, bv, (short)0, acc[ng], false, false);
    }
#pragma unroll
    for (int ng = 0; ng < 4; ng++) {
      v16h bv = __builtin_bit_cast(v16h, bfr[4 + ng]);
      acc[ng] = __builtin_amdgcn_wmma_f32_16x16x32_f16(
          false, av1, false, bv, (short)0, acc[ng], false, false);
    }

    a4 = a4n;
    lb = lbn;
    p ^= 1;   // next chunk stages the other buffer (no WAR with readers)
  }

  // ---- epilogue: D tile -> out (row-major fp32) ----
#pragma unroll
  for (int ng = 0; ng < 4; ng++) {
    int n = NB0 + ng * 16 + l16;
#pragma unroll
    for (int r = 0; r < 8; r++) {
      int m = MB + 8 * half + r;
      out[(size_t)m * OUTF + n] = acc[ng][r];
    }
  }
}

// ---------------------------------------------------------------------------
extern "C" void kernel_launch(void* const* d_in, const int* in_sizes, int n_in,
                              void* d_out, int out_size, void* d_ws, size_t ws_size,
                              hipStream_t stream) {
  const float* X   = (const float*)d_in[0];   // [8192, 512] f32
  const float* adj = (const float*)d_in[1];   // [8192, 8192] f32
  const int*   lab = (const int*)d_in[2];     // [8192] i32
  const float* W   = (const float*)d_in[3];   // [8, 512, 512] f32
  float* out = (float*)d_out;                 // [8192, 512] f32

  _Float16* WT   = (_Float16*)d_ws;                                   // 4 MB
  _Float16* supT = (_Float16*)((char*)d_ws +
                               (size_t)NSUB * INF * OUTF * sizeof(_Float16));  // 8 MB

  wt_transpose_kernel<<<(NSUB * INF * OUTF) / 256, 256, 0, stream>>>(W, WT);
  support_kernel<<<NN / 16, 256, 0, stream>>>(X, lab, WT, supT);
  gnn_kernel<<<NN / 16, 256, 0, stream>>>(adj, lab, supT, out);
}
